// Attention_51977694216863
// MI455X (gfx1250) — matrix operand order repaired and checked by reading, not compiled, not run on hardware
//
#include <hip/hip_runtime.h>
#include <stdint.h>

// ---------------------------------------------------------------------------
// Causal single-head attention for gfx1250 (MI455X), bf16 WMMA pipeline.
// B=4, S=2048, D=2048.  Workspace ~192 MiB:
//   qb,kb (bf16 row-major) | vbT (bf16, transposed: [D][B*S]) |
//   scores (f32) | attn (bf16) | ctx aliases qb.
// All GEMMs: 128x128x64 tiles, double-buffered dynamic LDS (73.7KB of the
// 320KB WGP pool), TDM (tensor_load_to_lds) staging for bf16 operands
// overlapped with WMMA compute; fp32 operands register-pipelined + converted.
// ---------------------------------------------------------------------------

#define BM 128
#define BN 128
#define BK 64
#define LDS_PITCH 72   // BK + 8 pad halfwords; 144B rows keep b128 LDS ops 16B-aligned
#define STAGE_ELEMS ((BM + BN) * LDS_PITCH)          // halfwords per stage
#define SMEM_BYTES  (2 * STAGE_ELEMS * 2)            // 73,728 B double-buffered

typedef __attribute__((ext_vector_type(16))) __bf16 v16bf;
typedef __attribute__((ext_vector_type(8)))  float  v8f;
typedef unsigned int u32x4 __attribute__((ext_vector_type(4)));
typedef int          i32x8 __attribute__((ext_vector_type(8)));
typedef int          i32x4 __attribute__((ext_vector_type(4)));

struct FragBF  { uint32_t u[8]; };
struct TileRegs { uint32_t w[2][8]; };   // one 128x64 tile as bf16, 32 elems/thread

static __device__ __forceinline__ uint16_t f2bf(float f) {
  return __builtin_bit_cast(uint16_t, (__bf16)f);   // native cvt, RNE
}
static __device__ __forceinline__ uint32_t pack2bf(float lo, float hi) {
  return (uint32_t)f2bf(lo) | ((uint32_t)f2bf(hi) << 16);
}

// ---------------------------------------------------------------------------
// Tensor Data Mover: 128-row x 64-elem bf16 tile (row-major, stride ld_elems)
// -> LDS at byte offset lds_off; 16B pad per 128B row => pitch 72 halfwords.
// Issued by one wave; caller waits TENSORcnt before use.
// ---------------------------------------------------------------------------
static __device__ __forceinline__ void tdm_load_tile_bf16(uint32_t lds_off,
                                                          const void* gsrc,
                                                          uint32_t ld_elems) {
  const uint64_t ga = (uint64_t)(uintptr_t)gsrc;
  u32x4 g0;
  g0[0] = 1u;                                   // count=1 (valid), user mode
  g0[1] = lds_off;                              // lds_addr (bytes)
  g0[2] = (uint32_t)ga;                         // global_addr[31:0]
  g0[3] = (uint32_t)(ga >> 32) | (2u << 30);    // global_addr[56:32] | type=2

  i32x8 g1;
  g1[0] = (int)((1u << 16)      // data_size = 1 -> 2 bytes
              | (1u << 20)      // pad_enable
              | (4u << 22)      // pad_interval: 32 DWORDs (one 64-elem row)
              | (3u << 25));    // pad_amount: 4 DWORDs (16B -> pitch 72)
  const uint32_t td0 = 1u << 20, td1 = 1u << 20;  // huge dims: exact-fit tiles, no OOB
  g1[1] = (int)((td0 & 0xFFFFu) << 16);
  g1[2] = (int)(((td0 >> 16) & 0xFFFFu) | ((td1 & 0xFFFFu) << 16));
  g1[3] = (int)(((td1 >> 16) & 0xFFFFu) | ((uint32_t)BK << 16));   // tile_dim0 = 64
  g1[4] = (int)(uint32_t)BM;                                       // tile_dim1 = 128
  g1[5] = (int)ld_elems;                                           // dim0_stride[31:0]
  g1[6] = 0;
  g1[7] = 0;

  i32x4 z4 = {0, 0, 0, 0};
#if defined(__clang_major__) && (__clang_major__ >= 23)
  i32x8 z8 = {0, 0, 0, 0, 0, 0, 0, 0};
  __builtin_amdgcn_tensor_load_to_lds(g0, g1, z4, z4, z8, 0);
#else
  __builtin_amdgcn_tensor_load_to_lds(g0, g1, z4, z4, 0);
#endif
}

// Load 16 contiguous source elements, produce 16 packed bf16 (8 dwords).
static __device__ __forceinline__ void load16_to_bf(const float* p, uint32_t w[8]) {
  const float4* q = (const float4*)p;
  float4 f0 = q[0], f1 = q[1], f2 = q[2], f3 = q[3];
  w[0] = pack2bf(f0.x, f0.y); w[1] = pack2bf(f0.z, f0.w);
  w[2] = pack2bf(f1.x, f1.y); w[3] = pack2bf(f1.z, f1.w);
  w[4] = pack2bf(f2.x, f2.y); w[5] = pack2bf(f2.z, f2.w);
  w[6] = pack2bf(f3.x, f3.y); w[7] = pack2bf(f3.z, f3.w);
}
static __device__ __forceinline__ void load16_to_bf(const uint16_t* p, uint32_t w[8]) {
  const uint4* q = (const uint4*)p;
  uint4 a = q[0], b = q[1];
  w[0] = a.x; w[1] = a.y; w[2] = a.z; w[3] = a.w;
  w[4] = b.x; w[5] = b.y; w[6] = b.z; w[7] = b.w;
}

// Register-pipelined staging: 128x64 tile -> regs (overlaps compute), then LDS.
template <typename T>
static __device__ __forceinline__ void load_tile_to_regs(const T* __restrict__ src,
                                                         int ld, TileRegs& tr) {
#pragma unroll
  for (int c = 0; c < 2; ++c) {
    const int idx = threadIdx.x + c * 256;       // 512 chunks of 16
    const int row = idx >> 2, q = idx & 3;
    load16_to_bf(src + (size_t)row * ld + q * 16, tr.w[c]);
  }
}
static __device__ __forceinline__ void store_tile_from_regs(const TileRegs& tr,
                                                            uint16_t* lds) {
#pragma unroll
  for (int c = 0; c < 2; ++c) {
    const int idx = threadIdx.x + c * 256;
    const int row = idx >> 2, q = idx & 3;
    uint32_t* d = (uint32_t*)(lds + row * LDS_PITCH + q * 16);
    ((uint4*)d)[0] = make_uint4(tr.w[c][0], tr.w[c][1], tr.w[c][2], tr.w[c][3]);
    ((uint4*)d)[1] = make_uint4(tr.w[c][4], tr.w[c][5], tr.w[c][6], tr.w[c][7]);
  }
}

template <typename T>
static __device__ __forceinline__ void prefetch_tile(const T* __restrict__ src, int ld) {
  const int idx = threadIdx.x;                   // warm a future 128x64 tile
  const int row = idx >> 1, half = idx & 1;
  __builtin_prefetch(src + (size_t)row * ld + half * 32, 0, 1);
}

// A fragment (16x32 bf16 at column kofs): lane = row m%16; VGPR0-3: K=kofs+h*8..,
// VGPR4-7: K=kofs+16+h*8..   (CDNA5 ISA §7.12.2)
static __device__ __forceinline__ v16bf load_frag_a(const uint16_t* lds, int row0, int kofs) {
  const int lane = threadIdx.x & 31;
  const int h = lane >> 4;
  const uint16_t* base = lds + (row0 + (lane & 15)) * LDS_PITCH + kofs;
  uint4 c0 = *(const uint4*)(base + h * 8);
  uint4 c1 = *(const uint4*)(base + 16 + h * 8);
  FragBF f;
  f.u[0] = c0.x; f.u[1] = c0.y; f.u[2] = c0.z; f.u[3] = c0.w;
  f.u[4] = c1.x; f.u[5] = c1.y; f.u[6] = c1.z; f.u[7] = c1.w;
  return __builtin_bit_cast(v16bf, f);
}
// B fragment (32x16 bf16): lane = col n%16; VGPR0-3: K=kofs+h*8.., VGPR4-7: K=kofs+16+h*8..
static __device__ __forceinline__ v16bf load_frag_b(const uint16_t* lds, int col0, int kofs) {
  const int lane = threadIdx.x & 31;
  const uint16_t* base = lds + (col0 + (lane & 15)) * LDS_PITCH + kofs + (lane >> 4) * 8;
  uint4 c0 = *(const uint4*)(base);
  uint4 c1 = *(const uint4*)(base + 16);
  FragBF f;
  f.u[0] = c0.x; f.u[1] = c0.y; f.u[2] = c0.z; f.u[3] = c0.w;
  f.u[4] = c1.x; f.u[5] = c1.y; f.u[6] = c1.z; f.u[7] = c1.w;
  return __builtin_bit_cast(v16bf, f);
}

template <typename T> static __device__ __forceinline__ void storeT(T* p, float v);
template <> __device__ __forceinline__ void storeT<float>(float* p, float v) { *p = v; }
template <> __device__ __forceinline__ void storeT<uint16_t>(uint16_t* p, float v) { *p = f2bf(v); }

// ---------------------------------------------------------------------------
// C[m,n] = sum_k A[m,k] * B[n,k]   (A * B^T), operands row-major.
// MODE 0: C[m,n]  = acc + bias[n]                (row-major store)
// MODE 1: C[m,n]  = acc*scale, causal mask       (scores, fp32 out)
// MODE 2: C[n,m]  = acc + bias[n]                (transposed bf16 store, packed b128)
// TDMA/TDMB: stage that operand with the Tensor Data Mover (bf16 only).
// KLIM: causal K-limit (kEnd = m0+BM) for attn*V.
// Double-buffered dynamic LDS; TDM / global loads for tile k+1 overlap
// WMMA compute of tile k; one barrier per k-step.
// ---------------------------------------------------------------------------
template <typename TA, typename TB, typename TOut, int MODE, bool TDMA, bool TDMB, bool KLIM>
__global__ __launch_bounds__(256) void gemm_abt(
    const TA* __restrict__ A, const TB* __restrict__ Bm, const float* __restrict__ bias,
    TOut* __restrict__ C, int lda, int ldb, int ldc, int K,
    long long sA, long long sB, long long sC, float scale) {
  extern __shared__ uint16_t smem[];
  const int z = blockIdx.z;
  A += (size_t)z * sA; Bm += (size_t)z * sB; C += (size_t)z * sC;
  const int m0 = blockIdx.y * BM, n0 = blockIdx.x * BN;
  const int wid = threadIdx.x >> 5, lane = threadIdx.x & 31;
  const int wm = (wid >> 2) * 64;  // {0,64}
  const int wn = (wid & 3) * 32;   // {0,32,64,96}

  if (MODE == 1 && n0 > m0 + (BM - 1)) {  // block fully above diagonal
    for (int i = 0; i < 4; ++i)
      for (int j = 0; j < 2; ++j) {
        const int nn = n0 + wn + j * 16 + (lane & 15);
        const int mb = m0 + wm + i * 16 + 8 * (lane >> 4);
        for (int r = 0; r < 8; ++r)
          storeT<TOut>(C + (size_t)(mb + r) * ldc + nn, -__builtin_inff());
      }
    return;
  }

  const int kEnd = KLIM ? min(K, m0 + BM) : K;
  const bool anyTDM = TDMA || TDMB;

  auto bufA = [&](int s) { return smem + s * STAGE_ELEMS; };
  auto bufB = [&](int s) { return smem + s * STAGE_ELEMS + BM * LDS_PITCH; };

  v8f acc[4][2];
#pragma unroll
  for (int i = 0; i < 4; ++i)
#pragma unroll
    for (int j = 0; j < 2; ++j)
#pragma unroll
      for (int r = 0; r < 8; ++r) acc[i][j][r] = 0.0f;

  // ---- prologue: stage k0=0 into buffer 0 ----
  {
    if (anyTDM && threadIdx.x < 32) {
      if (TDMA) tdm_load_tile_bf16((uint32_t)(uintptr_t)(void*)bufA(0),
                                   (const void*)(A + (size_t)m0 * lda), lda);
      if (TDMB) tdm_load_tile_bf16((uint32_t)(uintptr_t)(void*)bufB(0),
                                   (const void*)(Bm + (size_t)n0 * ldb), ldb);
    }
    TileRegs ra, rb;
    if (!TDMA) load_tile_to_regs(A + (size_t)m0 * lda, lda, ra);
    if (!TDMB) load_tile_to_regs(Bm + (size_t)n0 * ldb, ldb, rb);
    if (!TDMA) store_tile_from_regs(ra, bufA(0));
    if (!TDMB) store_tile_from_regs(rb, bufB(0));
    if (anyTDM && threadIdx.x < 32) __builtin_amdgcn_s_wait_tensorcnt(0);
    __syncthreads();
  }

  int sb = 0;
  for (int k0 = 0; k0 < kEnd; k0 += BK, sb ^= 1) {
    const bool more = (k0 + BK) < kEnd;
    TileRegs ra, rb;
    if (more) {  // kick off tile k0+BK into the other buffer (overlaps compute)
      if (anyTDM && threadIdx.x < 32) {
        if (TDMA) tdm_load_tile_bf16((uint32_t)(uintptr_t)(void*)bufA(sb ^ 1),
                                     (const void*)(A + (size_t)m0 * lda + k0 + BK), lda);
        if (TDMB) tdm_load_tile_bf16((uint32_t)(uintptr_t)(void*)bufB(sb ^ 1),
                                     (const void*)(Bm + (size_t)n0 * ldb + k0 + BK), ldb);
      }
      if (!TDMA) load_tile_to_regs(A + (size_t)m0 * lda + k0 + BK, lda, ra);
      if (!TDMB) load_tile_to_regs(Bm + (size_t)n0 * ldb + k0 + BK, ldb, rb);
      if (!TDMA && k0 + 2 * BK < kEnd) prefetch_tile(A + (size_t)m0 * lda + k0 + 2 * BK, lda);
      if (!TDMB && k0 + 2 * BK < kEnd) prefetch_tile(Bm + (size_t)n0 * ldb + k0 + 2 * BK, ldb);
    }

    {  // compute on current buffer
      const uint16_t* la = bufA(sb);
      const uint16_t* lb = bufB(sb);
#pragma unroll
      for (int s = 0; s < 2; ++s) {
        v16bf af[4], bf[2];
#pragma unroll
        for (int i = 0; i < 4; ++i) af[i] = load_frag_a(la, wm + i * 16, s * 32);
#pragma unroll
        for (int j = 0; j < 2; ++j) bf[j] = load_frag_b(lb, wn + j * 16, s * 32);
#pragma unroll
        for (int i = 0; i < 4; ++i)
#pragma unroll
          for (int j = 0; j < 2; ++j)
            acc[i][j] = __builtin_amdgcn_wmma_f32_16x16x32_bf16(
                false, af[i], false, bf[j], (short)0, acc[i][j], false, false);
      }
    }

    if (more) {
      if (!TDMA) store_tile_from_regs(ra, bufA(sb ^ 1));
      if (!TDMB) store_tile_from_regs(rb, bufB(sb ^ 1));
      if (anyTDM && threadIdx.x < 32) __builtin_amdgcn_s_wait_tensorcnt(0);
      __syncthreads();
    }
  }

  // ---- epilogue ----
#pragma unroll
  for (int i = 0; i < 4; ++i)
#pragma unroll
    for (int j = 0; j < 2; ++j) {
      const int nn = n0 + wn + j * 16 + (lane & 15);
      const int mb = m0 + wm + i * 16 + 8 * (lane >> 4);
      const float bv = (MODE != 1 && bias) ? bias[nn] : 0.0f;
      if (MODE == 2) {
        // transposed store: lane holds 8 consecutive m's for column nn -> one b128
        uint32_t p[4];
#pragma unroll
        for (int r = 0; r < 4; ++r)
          p[r] = pack2bf(acc[i][j][2 * r] + bv, acc[i][j][2 * r + 1] + bv);
        *(uint4*)((uint16_t*)C + (size_t)nn * ldc + mb) = make_uint4(p[0], p[1], p[2], p[3]);
      } else {
#pragma unroll
        for (int r = 0; r < 8; ++r) {
          const int mm = mb + r;
          float v = acc[i][j][r] * scale + bv;
          if (MODE == 1 && nn > mm) v = -__builtin_inff();
          storeT<TOut>(C + (size_t)mm * ldc + nn, v);
        }
      }
    }
}

// ---------------------------------------------------------------------------
// Row softmax: one 256-thread block per row of 2048 fp32 scores -> bf16 attn.
// ---------------------------------------------------------------------------
__global__ __launch_bounds__(256) void softmax_row(const float* __restrict__ scores,
                                                   uint16_t* __restrict__ attn, int S) {
  const size_t row = blockIdx.x;
  const float* src = scores + row * (size_t)S;
  uint16_t* dst = attn + row * (size_t)S;
  const int t = threadIdx.x;
  float v[8];
#pragma unroll
  for (int i = 0; i < 8; ++i) v[i] = src[t + i * 256];

  __shared__ float red[8];
  float m = v[0];
#pragma unroll
  for (int i = 1; i < 8; ++i) m = fmaxf(m, v[i]);
  for (int off = 16; off > 0; off >>= 1) m = fmaxf(m, __shfl_xor(m, off, 32));
  if ((t & 31) == 0) red[t >> 5] = m;
  __syncthreads();
  float bm = red[0];
#pragma unroll
  for (int w = 1; w < 8; ++w) bm = fmaxf(bm, red[w]);
  __syncthreads();

  float s = 0.0f;
#pragma unroll
  for (int i = 0; i < 8; ++i) { v[i] = __expf(v[i] - bm); s += v[i]; }
  for (int off = 16; off > 0; off >>= 1) s += __shfl_xor(s, off, 32);
  if ((t & 31) == 0) red[t >> 5] = s;
  __syncthreads();
  float bs = 0.0f;
#pragma unroll
  for (int w = 0; w < 8; ++w) bs += red[w];
  const float inv = 1.0f / bs;
#pragma unroll
  for (int i = 0; i < 8; ++i) dst[t + i * 256] = f2bf(v[i] * inv);
}

// ---------------------------------------------------------------------------
extern "C" void kernel_launch(void* const* d_in, const int* in_sizes, int n_in,
                              void* d_out, int out_size, void* d_ws, size_t ws_size,
                              hipStream_t stream) {
  (void)in_sizes; (void)n_in; (void)out_size; (void)ws_size;
  const int Bb = 4, S = 2048, D = 2048;
  const int BbS = Bb * S;                      // 8192
  const long long SD = (long long)Bb * S * D;  // 16,777,216 (== B*S*S since S==D)

  const float* x  = (const float*)d_in[0];
  // d_in[1] = mask (causal; implied analytically, not read)
  const float* Wq = (const float*)d_in[2];
  const float* bq = (const float*)d_in[3];
  const float* Wk = (const float*)d_in[4];
  const float* bk = (const float*)d_in[5];
  const float* Wv = (const float*)d_in[6];
  const float* bv = (const float*)d_in[7];
  const float* Wp = (const float*)d_in[8];
  const float* bp = (const float*)d_in[9];
  float* out = (float*)d_out;

  uint16_t* qb  = (uint16_t*)d_ws;             // 32 MiB bf16 [B*S, D]
  uint16_t* kb  = qb + SD;                     // 32 MiB bf16 [B*S, D]
  uint16_t* vbT = kb + SD;                     // 32 MiB bf16 [D, B*S] (transposed V)
  float* scores = (float*)(vbT + SD);          // 64 MiB fp32 [B, S, S]
  uint16_t* attn = (uint16_t*)(scores + SD);   // 32 MiB bf16 [B, S, S]
  uint16_t* ctx = qb;                          // reuse q region (dead after scores)

  const dim3 blk(256);
  const dim3 gproj(D / BN, BbS / BM, 1);       // (16, 64)
  const dim3 gatt(S / BN, S / BM, Bb);         // (16, 16, 4)
  const float inv_std = 0.02209708691207961f;  // 1/sqrt(2048)

  // q,k projections: [8192,2048] x [2048,2048]^T -> bf16 row-major
  gemm_abt<float, float, uint16_t, 0, false, false, false><<<gproj, blk, SMEM_BYTES, stream>>>(
      x, Wq, bq, qb, D, D, D, D, 0, 0, 0, 1.0f);
  gemm_abt<float, float, uint16_t, 0, false, false, false><<<gproj, blk, SMEM_BYTES, stream>>>(
      x, Wk, bk, kb, D, D, D, D, 0, 0, 0, 1.0f);
  // v projection stored TRANSPOSED: vbT[e, b*S+s]  (packed b128 stores)
  gemm_abt<float, float, uint16_t, 2, false, false, false><<<gproj, blk, SMEM_BYTES, stream>>>(
      x, Wv, bv, vbT, D, D, BbS, D, 0, 0, 0, 1.0f);

  // scores = q k^T * inv_std, causal mask, per batch -> fp32 (TDM-staged tiles)
  gemm_abt<uint16_t, uint16_t, float, 1, true, true, false><<<gatt, blk, SMEM_BYTES, stream>>>(
      qb, kb, nullptr, scores, D, D, S, D,
      (long long)S * D, (long long)S * D, (long long)S * S, inv_std);

  // row softmax -> bf16 attn
  softmax_row<<<dim3(Bb * S), blk, 0, stream>>>(scores, attn, S);

  // ctx = attn @ v = attn * (vT)^T, causal K-limit, per batch -> bf16 (TDM both)
  gemm_abt<uint16_t, uint16_t, uint16_t, 0, true, true, true><<<gatt, blk, SMEM_BYTES, stream>>>(
      attn, vbT, nullptr, ctx, S, BbS, D, S,
      (long long)S * S, (long long)S, (long long)S * D, 1.0f);

  // out = ctx Wp^T + bp -> fp32 (TDM-staged A)
  gemm_abt<uint16_t, float, float, 0, true, false, false><<<gproj, blk, SMEM_BYTES, stream>>>(
      ctx, Wp, bp, out, D, D, D, D, 0, 0, 0, 1.0f);
}
